// GatedGCN_81509889343768
// MI455X (gfx1250) — matrix operand order repaired and checked by reading, not compiled, not run on hardware
//
#include <hip/hip_runtime.h>
#include <hip/hip_bf16.h>

typedef __attribute__((ext_vector_type(2)))  float    v2f;
typedef __attribute__((ext_vector_type(8)))  float    v8f;
typedef __attribute__((ext_vector_type(16))) _Float16 v16h;

#define N_NODES 100000
#define N_EDGES 1600000
#define DIM     128
#define OUTD    64
#define MD      16
#define GH      64
#define LN_EPS  1e-5f
#define GATE_TPW 10   // edge-tiles (of 16) per wave; 1250 blocks * 8 waves * 160 = E

// |x| via raw VALU bit-and: opaque to InstCombine so it can never be
// re-canonicalized into llvm.fabs (whose v2f32 lowering is broken on this
// gfx1250 toolchain).
__device__ __forceinline__ float abs_asm(float x) {
    float r;
    asm("v_and_b32 %0, 0x7fffffff, %1" : "=v"(r) : "v"(x));
    return r;
}

// ---------------------------------------------------------------------------
// Kernel 1: Edge gate MLP.  wave = 16 edges/tile, 10 tiles per wave.
// h = relu(feats@W1 + b1) via v_wmma_f32_16x16x32_f16 (K=64 -> 2 wmma per
// 16-col tile, 4 tiles), then g = h @ w2, sigmoid.
// W1 is staged in LDS already permuted into B-fragment order so each lane
// reads its v16h fragment as one contiguous 32 B LDS access.
// ---------------------------------------------------------------------------
__global__ __launch_bounds__(256)
void gate_mlp_kernel(const float* __restrict__ motif, const int* __restrict__ ei,
                     const float* __restrict__ w1, const float* __restrict__ b1,
                     const float* __restrict__ w2, const float* __restrict__ b2,
                     float* __restrict__ gate)
{
    // w1p[((j*2+t)*2+hi)*256 + l16*16 + q] = w1[(32t+8hi+q+(q&8))*GH + 16j+l16]
    __shared__ __align__(32) _Float16 w1p[4096];   // 8 KB
    __shared__ float red[8][2][16][8];             // [wave][half][lane16][v] 8 KB

    const int tid  = threadIdx.x;
    const int wave = tid >> 5;
    const int lane = tid & 31;
    const int l16  = lane & 15;
    const int hi   = lane >> 4;            // 0 = lanes 0-15, 1 = lanes 16-31

    for (int i = tid; i < 4096; i += 256) {
        const int q  = i & 15;
        const int ls = (i >> 4) & 15;
        const int hs = (i >> 8) & 1;
        const int ts = (i >> 9) & 1;
        const int js = (i >> 10) & 3;
        w1p[i] = (_Float16)w1[(32 * ts + 8 * hs + q + (q & 8)) * GH + 16 * js + ls];
    }
    __syncthreads();

    // B fragments: f16 B 32x16 layout. lane<16: local K {0..7,16..23}, lane>=16:
    // K {8..15,24..31}; halves[q] = K = 8*hi+q+(q&8) for column N = 16*j + l16.
    v16h Bf[4][2];
    #pragma unroll
    for (int j = 0; j < 4; ++j)
        #pragma unroll
        for (int t = 0; t < 2; ++t)
            Bf[j][t] = *(const v16h*)(w1p + (((j * 2 + t) * 2 + hi) << 8) + (l16 << 4));

    float b1u[4], w2u[4];
    #pragma unroll
    for (int j = 0; j < 4; ++j) {
        b1u[j] = b1[16 * j + l16];
        w2u[j] = w2[16 * j + l16];
    }
    const float b2v = b2[0];

    const long eWave = ((long)blockIdx.x * 8 + wave) * (16 * GATE_TPW);
    const int  off   = hi ? 8 : 0;         // which half of the 16 motif feats

    for (int tile = 0; tile < GATE_TPW; ++tile) {
        // A fragments: f16 A 16x32 layout. lane holds row m=l16; lane<16 ->
        // K {0..7,16..23}, lane>=16 -> K {8..15,24..31}.
        // feats K-map: [0..15]=mu, [16..31]=mv, [32..47]=|mu-mv|, [48..63]=mu*mv
        const long e   = eWave + tile * 16 + l16;
        const int  src = ei[e];
        const int  dst = ei[N_EDGES + e];

        float mu[8], mv[8];
        #pragma unroll
        for (int q = 0; q < 8; ++q) {
            mu[q] = motif[(long)src * MD + off + q];
            mv[q] = motif[(long)dst * MD + off + q];
        }
        v16h A0, A1;
        #pragma unroll
        for (int q = 0; q < 8; ++q) {
            A0[q]     = (_Float16)mu[q];
            A0[q + 8] = (_Float16)mv[q];
            A1[q]     = (_Float16)abs_asm(mu[q] - mv[q]);
            A1[q + 8] = (_Float16)(mu[q] * mv[q]);
        }

        float partial[8];
        #pragma unroll
        for (int v = 0; v < 8; ++v) partial[v] = 0.f;

        #pragma unroll
        for (int j = 0; j < 4; ++j) {
            v8f c = {};
            c = __builtin_amdgcn_wmma_f32_16x16x32_f16(false, A0, false, Bf[j][0],
                                                       (short)0, c, false, false);
            c = __builtin_amdgcn_wmma_f32_16x16x32_f16(false, A1, false, Bf[j][1],
                                                       (short)0, c, false, false);
            #pragma unroll
            for (int v = 0; v < 8; ++v) {
                float h = c[v] + b1u[j];        // hidden unit 16j+l16, row v+hi*8
                h = h > 0.f ? h : 0.f;
                partial[v] += h * w2u[j];
            }
        }

        // reduce each row's dot product across the 16 lanes of its half
        __syncthreads();                        // WAR vs previous tile's reads
        #pragma unroll
        for (int v = 0; v < 8; ++v) red[wave][hi][l16][v] = partial[v];
        __syncthreads();

        if (hi == 0) {                          // lane l16 emits edge row m=l16
            const int hr = l16 >> 3, vr = l16 & 7;
            float g = 0.f;
            #pragma unroll
            for (int l = 0; l < 16; ++l) g += red[wave][hr][l][vr];
            g += b2v;
            float s = 1.f / (1.f + __expf(-g)); // TAU = 1
            s = fminf(fmaxf(s, 0.f), 1.f);
            gate[eWave + tile * 16 + l16] = s;
        }
    }
}

// ---------------------------------------------------------------------------
// Degree / norm kernels
// ---------------------------------------------------------------------------
__global__ void deg_init_kernel(float* __restrict__ deg)
{
    int n = blockIdx.x * 256 + threadIdx.x;
    if (n < N_NODES) deg[n] = 1.0f;             // self-loop weight
}

__global__ void deg_scatter_kernel(const int* __restrict__ ei,
                                   const float* __restrict__ gate,
                                   float* __restrict__ deg)
{
    long e = (long)blockIdx.x * 256 + threadIdx.x;
    if (e < N_EDGES) atomicAdd(&deg[ei[N_EDGES + e]], gate[e]);
}

__global__ void dinv_kernel(float* __restrict__ deg)
{
    int n = blockIdx.x * 256 + threadIdx.x;
    if (n < N_NODES) {
        float d = deg[n];
        deg[n] = d > 0.f ? rsqrtf(d) : 0.f;     // in place: deg -> dinv
    }
}

__global__ void norm_kernel(const int* __restrict__ ei,
                            const float* __restrict__ dinv,
                            float* __restrict__ gate)  // in place: gate -> norm
{
    long e = (long)blockIdx.x * 256 + threadIdx.x;
    if (e < N_EDGES)
        gate[e] = dinv[ei[e]] * gate[e] * dinv[ei[N_EDGES + e]];
}

// ---------------------------------------------------------------------------
// Tiled fp32 WMMA GEMM: Y[N_NODES,NOUT] = X[N_NODES,128] @ W[128,NOUT] (+bias)
// block = 256 thr = 8 waves, 128 nodes per block; v_wmma_f32_16x16x4_f32.
// ---------------------------------------------------------------------------
template <int NOUT, bool ADD_BIAS>
__global__ __launch_bounds__(256)
void node_gemm_kernel(const float* __restrict__ X, const float* __restrict__ W,
                      const float* __restrict__ bias, float* __restrict__ Y)
{
    __shared__ float xs[128 * 16];              // 128 nodes x 16 k  (8 KB)
    __shared__ float wsh[16 * NOUT];            // 16 k x NOUT       (<= 8 KB)

    const int tid  = threadIdx.x;
    const int wave = tid >> 5;
    const int lane = tid & 31;
    const int l16  = lane & 15;
    const int hi   = lane >> 4;
    const int nodeBase = blockIdx.x * 128;
    constexpr int NT = NOUT / 16;

    v8f acc[NT] = {};

    for (int kc = 0; kc < DIM; kc += 16) {
        __syncthreads();
        for (int i = tid; i < 128 * 16; i += 256) {
            int node = nodeBase + (i >> 4);
            xs[i] = (node < N_NODES) ? X[(long)node * DIM + kc + (i & 15)] : 0.f;
        }
        for (int i = tid; i < 16 * NOUT; i += 256) {
            int kk = i / NOUT, n = i % NOUT;
            wsh[i] = W[(long)(kc + kk) * NOUT + n];
        }
        __syncthreads();
        #pragma unroll
        for (int ks = 0; ks < 4; ++ks) {
            // f32 A 16x4: lanes 0-15 K={0,1}, lanes 16-31 K={2,3}; B matches.
            const int ka = ks * 4 + (hi ? 2 : 0);
            v2f a;
            a.x = xs[(wave * 16 + l16) * 16 + ka];
            a.y = xs[(wave * 16 + l16) * 16 + ka + 1];
            #pragma unroll
            for (int j = 0; j < NT; ++j) {
                v2f b;
                b.x = wsh[ka * NOUT + 16 * j + l16];
                b.y = wsh[(ka + 1) * NOUT + 16 * j + l16];
                acc[j] = __builtin_amdgcn_wmma_f32_16x16x4_f32(
                    false, a, false, b, (short)0, acc[j], false, false);
            }
        }
    }

    // C/D layout: VGPR v -> row v (lanes 0-15) or v+8 (lanes 16-31), col = l16
    const int rowBase = nodeBase + wave * 16 + hi * 8;
    #pragma unroll
    for (int j = 0; j < NT; ++j) {
        const int col = 16 * j + l16;
        const float bb = ADD_BIAS ? bias[col] : 0.f;
        #pragma unroll
        for (int v = 0; v < 8; ++v) {
            const int node = rowBase + v;
            if (node < N_NODES) Y[(long)node * NOUT + col] = acc[j][v] + bb;
        }
    }
}

// ---------------------------------------------------------------------------
// hh init: self-loop term + bias.  hh[n,f] = dinv[n]^2 * xw[n,f] + conv_b[f]
// ---------------------------------------------------------------------------
__global__ void hh_init_kernel(const float* __restrict__ xw,
                               const float* __restrict__ dinv,
                               const float* __restrict__ cb,
                               float* __restrict__ hh)
{
    long i = (long)blockIdx.x * 256 + threadIdx.x;
    if (i < (long)N_NODES * DIM) {
        int n = (int)(i >> 7), f = (int)(i & 127);
        float di = dinv[n];
        hh[i] = di * di * xw[i] + cb[f];
    }
}

// ---------------------------------------------------------------------------
// Edge scatter: wave per edge, float4 per lane, global f32 atomics into hh[dst]
// ---------------------------------------------------------------------------
__global__ __launch_bounds__(256)
void edge_scatter_kernel(const int* __restrict__ ei, const float* __restrict__ nrm,
                         const float* __restrict__ xw, float* __restrict__ hh)
{
    const long e = (((long)blockIdx.x * 256) + threadIdx.x) >> 5;
    const int lane = threadIdx.x & 31;
    if (e >= N_EDGES) return;
    const int src = ei[e], dst = ei[N_EDGES + e];
    const float w = nrm[e];
    const float4 v = *(const float4*)(xw + (long)src * DIM + lane * 4);
    float* h = hh + (long)dst * DIM + lane * 4;
    atomicAdd(h + 0, w * v.x);
    atomicAdd(h + 1, w * v.y);
    atomicAdd(h + 2, w * v.z);
    atomicAdd(h + 3, w * v.w);
}

// ---------------------------------------------------------------------------
// LayerNorm + ReLU + residual: wave per node, wave32 shuffle reduction
// ---------------------------------------------------------------------------
__global__ __launch_bounds__(256)
void ln_residual_kernel(const float* __restrict__ xin, const float* __restrict__ hh,
                        const float* __restrict__ g, const float* __restrict__ b,
                        float* __restrict__ xout)
{
    const long node = (((long)blockIdx.x * 256) + threadIdx.x) >> 5;
    const int lane = threadIdx.x & 31;
    if (node >= N_NODES) return;
    const int f = lane * 4;
    const float4 h4 = *(const float4*)(hh + node * DIM + f);
    float s  = h4.x + h4.y + h4.z + h4.w;
    float s2 = h4.x * h4.x + h4.y * h4.y + h4.z * h4.z + h4.w * h4.w;
    #pragma unroll
    for (int m = 1; m < 32; m <<= 1) {
        s  += __shfl_xor(s,  m, 32);
        s2 += __shfl_xor(s2, m, 32);
    }
    const float mean = s * (1.f / DIM);
    const float var  = s2 * (1.f / DIM) - mean * mean;
    const float rs   = rsqrtf(var + LN_EPS);
    const float4 x4  = *(const float4*)(xin + node * DIM + f);
    float4 o;
    o.x = x4.x + fmaxf((h4.x - mean) * rs * g[f + 0] + b[f + 0], 0.f);
    o.y = x4.y + fmaxf((h4.y - mean) * rs * g[f + 1] + b[f + 1], 0.f);
    o.z = x4.z + fmaxf((h4.z - mean) * rs * g[f + 2] + b[f + 2], 0.f);
    o.w = x4.w + fmaxf((h4.w - mean) * rs * g[f + 3] + b[f + 3], 0.f);
    *(float4*)(xout + node * DIM + f) = o;
}

// ---------------------------------------------------------------------------
extern "C" void kernel_launch(void* const* d_in, const int* in_sizes, int n_in,
                              void* d_out, int out_size, void* d_ws, size_t ws_size,
                              hipStream_t stream)
{
    const float* x     = (const float*)d_in[0];
    const float* motif = (const float*)d_in[1];
    const int*   ei    = (const int*)  d_in[2];
    const float* gw1   = (const float*)d_in[3];
    const float* gb1   = (const float*)d_in[4];
    const float* gw2   = (const float*)d_in[5];
    const float* gb2   = (const float*)d_in[6];
    const float* convw = (const float*)d_in[7];
    const float* convb = (const float*)d_in[8];
    const float* lng   = (const float*)d_in[9];
    const float* lnb   = (const float*)d_in[10];
    const float* hw    = (const float*)d_in[11];
    const float* hb    = (const float*)d_in[12];
    float* out = (float*)d_out;

    float* ws   = (float*)d_ws;
    float* gate = ws;  ws += N_EDGES;                 // gate -> norm (in place)
    float* deg  = ws;  ws += N_NODES;                 // deg  -> dinv (in place)
    float* xw   = ws;  ws += (size_t)N_NODES * DIM;
    float* hh   = ws;  ws += (size_t)N_NODES * DIM;
    float* xcur = ws;  ws += (size_t)N_NODES * DIM;

    // 1) edge gates (WMMA f16 MLP): 1250 blocks * 8 waves * 160 edges = E
    gate_mlp_kernel<<<N_EDGES / (128 * GATE_TPW), 256, 0, stream>>>(
        motif, ei, gw1, gb1, gw2, gb2, gate);
    // 2) gcn_norm with self loops
    deg_init_kernel<<<(N_NODES + 255) / 256, 256, 0, stream>>>(deg);
    deg_scatter_kernel<<<(N_EDGES + 255) / 256, 256, 0, stream>>>(ei, gate, deg);
    dinv_kernel<<<(N_NODES + 255) / 256, 256, 0, stream>>>(deg);
    norm_kernel<<<(N_EDGES + 255) / 256, 256, 0, stream>>>(ei, deg, gate);

    // 3) two GCN layers
    const float* xin = x;
    const long nf = (long)N_NODES * DIM;
    for (int layer = 0; layer < 2; ++layer) {
        node_gemm_kernel<DIM, false><<<(N_NODES + 127) / 128, 256, 0, stream>>>(
            xin, convw + (long)layer * DIM * DIM, nullptr, xw);
        hh_init_kernel<<<(int)((nf + 255) / 256), 256, 0, stream>>>(
            xw, deg, convb + layer * DIM, hh);
        edge_scatter_kernel<<<(int)(((long)N_EDGES * 32 + 255) / 256), 256, 0,
                              stream>>>(ei, gate, xw, hh);
        ln_residual_kernel<<<(int)(((long)N_NODES * 32 + 255) / 256), 256, 0,
                             stream>>>(xin, hh, lng + layer * DIM,
                                       lnb + layer * DIM, xcur);
        xin = xcur;
    }

    // 4) head
    node_gemm_kernel<OUTD, true><<<(N_NODES + 127) / 128, 256, 0, stream>>>(
        xin, hw, hb, out);
}
